// PositionalEmbedder_18236431139168
// MI455X (gfx1250) — matrix operand order repaired and checked by reading, not compiled, not run on hardware
//
#include <hip/hip_runtime.h>
#include <math.h>

// ---------------- types ----------------
typedef _Float16 half_t;
typedef __attribute__((ext_vector_type(16))) _Float16 v16h;
typedef __attribute__((ext_vector_type(8)))  float    v8f;
typedef __attribute__((ext_vector_type(4)))  int      v4i_t;

#define GLOBAL_AS __attribute__((address_space(1)))
#define LDS_AS    __attribute__((address_space(3)))

#define HID 768
#define HEADS 12
#define DH 64
#define NL 4
#define BATCH 8
#define SEQ 511
#define LFULL 512
#define ROWS (BATCH * LFULL)   // 4096
#define FFN (4 * HID)          // 3072
#define EPS 1e-7f
#define INV_SCALE 0.0721687836487032f  // 1/sqrt(3*64)
#define NEG_INF (-3.402823466e38f)

// CDNA5 async memory->LDS copy (ASYNCcnt path), with safe fallback.
#if defined(__has_builtin)
#if __has_builtin(__builtin_amdgcn_global_load_async_to_lds_b128) && \
    __has_builtin(__builtin_amdgcn_s_wait_asynccnt)
#define HAVE_ASYNC_LDS 1
#endif
#endif
#ifndef HAVE_ASYNC_LDS
#define HAVE_ASYNC_LDS 0
#endif

#if HAVE_ASYNC_LDS
// 16B global -> LDS async copy. Builtin signature (probe-confirmed via clang
// diagnostic): (v4i AS1*, v4i AS3*, imm offset, imm cpol).
__device__ __forceinline__ void async_copy16(const half_t* g, half_t* l) {
    GLOBAL_AS const v4i_t* gp = (GLOBAL_AS const v4i_t*)g;
    LDS_AS v4i_t* lp = (LDS_AS v4i_t*)l;
    __builtin_amdgcn_global_load_async_to_lds_b128(
        const_cast<GLOBAL_AS v4i_t*>(gp), lp, 0, 0);
}
#endif

// ---------------- block reductions (256 threads) ----------------
__device__ __forceinline__ float block_sum(float v, float* red) {
    int t = threadIdx.x;
    red[t] = v; __syncthreads();
    for (int s = 128; s > 0; s >>= 1) { if (t < s) red[t] += red[t + s]; __syncthreads(); }
    float r = red[0]; __syncthreads();
    return r;
}
__device__ __forceinline__ float block_max(float v, float* red) {
    int t = threadIdx.x;
    red[t] = v; __syncthreads();
    for (int s = 128; s > 0; s >>= 1) { if (t < s) red[t] = fmaxf(red[t], red[t + s]); __syncthreads(); }
    float r = red[0]; __syncthreads();
    return r;
}

// ---------------- WMMA GEMM: C[M,N] = A[M,K] @ Bt[N,K]^T (+bias, +gelu) ----------------
// A, Bt f16 row-major. 256 threads = 8 waves (4 M-strips x 2 N-strips).
// Each wave: 32 x (BNT/2) output tile = 2 x (BNT/32) wmma accumulators.
#define BM 128
#define BK 32
#define LDT 40  // padded LDS stride (halfs)

template <int BNT>
__global__ __launch_bounds__(256) void k_gemm(
    const half_t* __restrict__ A, const half_t* __restrict__ Bt,
    const float* __restrict__ bias, float* __restrict__ C, half_t* __restrict__ C16,
    int M, int N, int K, int lda, int ldb, int ldc,
    long sA, long sB, long sC, int gelu)
{
    constexpr int WN = BNT / 2;    // cols per wave
    constexpr int NF = WN / 16;    // B fragments per wave (2 or 4)

    __shared__ half_t As[BM][LDT];
    __shared__ half_t Bs[BNT][LDT];

    const int z = blockIdx.z;
    A  += (long)z * sA;
    Bt += (long)z * sB;
    if (C)   C   += (long)z * sC;
    if (C16) C16 += (long)z * sC;

    const int row0 = blockIdx.y * BM;
    const int col0 = blockIdx.x * BNT;
    const int tid  = threadIdx.x;
    const int lane = tid & 31;
    const int wid  = tid >> 5;        // 0..7
    const int wm   = wid >> 1;        // 0..3 -> 32-row strip
    const int wn   = wid & 1;         // 0..1 -> WN-col strip
    const int grp  = lane >> 4;       // lane group (WMMA layout)
    const int l15  = lane & 15;

    v8f acc[2][NF] = {};

    for (int k0 = 0; k0 < K; k0 += BK) {
        // stage A tile: 128x32 halfs = 512 16B chunks, 2 per thread
        #pragma unroll
        for (int i = 0; i < 2; i++) {
            int c = tid + i * 256;
            int r = c >> 2, col = (c & 3) * 8;
#if HAVE_ASYNC_LDS
            async_copy16(&A[(long)(row0 + r) * lda + k0 + col], &As[r][col]);
#else
            *(float4*)&As[r][col] =
                *(const float4*)&A[(long)(row0 + r) * lda + k0 + col];
#endif
        }
        // stage Bt tile: BNTx32 halfs, BNT/64 chunks per thread
        #pragma unroll
        for (int i = 0; i < BNT / 64; i++) {
            int c = tid + i * 256;
            int r = c >> 2, col = (c & 3) * 8;
#if HAVE_ASYNC_LDS
            async_copy16(&Bt[(long)(col0 + r) * ldb + k0 + col], &Bs[r][col]);
#else
            *(float4*)&Bs[r][col] =
                *(const float4*)&Bt[(long)(col0 + r) * ldb + k0 + col];
#endif
        }
#if HAVE_ASYNC_LDS
        __builtin_amdgcn_s_wait_asynccnt(0);
#else
        if (k0 + BK < K) {  // hint next tiles toward L2 (global_prefetch_b8)
            __builtin_prefetch(&A[(long)(row0 + (tid >> 1)) * lda + k0 + BK], 0, 1);
            __builtin_prefetch(&Bt[(long)(col0 + (tid >> 2)) * ldb + k0 + BK], 0, 1);
        }
#endif
        __syncthreads();

        // A fragments: 16x32, lane l15 = row, halfs 0..7 @ K=8*grp, 8..15 @ K=16+8*grp
        union Frag { v16h v; float4 f4[2]; };
        Frag af[2], bf[NF];
        #pragma unroll
        for (int i = 0; i < 2; i++) {
            int r = wm * 32 + i * 16 + l15;
            af[i].f4[0] = *(const float4*)&As[r][8 * grp];
            af[i].f4[1] = *(const float4*)&As[r][16 + 8 * grp];
        }
        // B fragments: 32x16, lane l15 = col (row of Bt), halfs 0..7 @ K=8*grp, 8..15 @ K=16+8*grp
        #pragma unroll
        for (int j = 0; j < NF; j++) {
            int r = wn * WN + j * 16 + l15;
            bf[j].f4[0] = *(const float4*)&Bs[r][8 * grp];
            bf[j].f4[1] = *(const float4*)&Bs[r][16 + 8 * grp];
        }
        #pragma unroll
        for (int i = 0; i < 2; i++)
            #pragma unroll
            for (int j = 0; j < NF; j++)
                acc[i][j] = __builtin_amdgcn_wmma_f32_16x16x32_f16(
                    false, af[i].v, false, bf[j].v, (short)0, acc[i][j], false, false);
        __syncthreads();
    }

    // epilogue: C/D layout -> VGPR r holds (M = r + 8*grp, N = l15)
    #pragma unroll
    for (int i = 0; i < 2; i++) {
        #pragma unroll
        for (int j = 0; j < NF; j++) {
            int col = col0 + wn * WN + j * 16 + l15;
            float bv = bias ? bias[col] : 0.0f;
            #pragma unroll
            for (int r = 0; r < 8; r++) {
                int row = row0 + wm * 32 + i * 16 + 8 * grp + r;
                float v = acc[i][j][r] + bv;
                if (gelu) v = 0.5f * v * (1.0f + erff(v * 0.70710678118f));
                long idx = (long)row * ldc + col;
                if (C)   C[idx]   = v;
                if (C16) C16[idx] = (half_t)v;
            }
        }
    }
}

// ---------------- embedding: cls concat + LN + pos_emb; positions_full ----------------
__global__ __launch_bounds__(256) void k_embed(
    const float* __restrict__ iv, const int* __restrict__ ip,
    const float* __restrict__ cls, const float* __restrict__ pos_emb,
    const float* __restrict__ s, const float* __restrict__ bb,
    float* __restrict__ seq, half_t* __restrict__ seqh, int* __restrict__ posf)
{
    __shared__ float red[256];
    const int row = blockIdx.x;          // b*512 + l
    const int b = row >> 9, l = row & 511;
    const int t = threadIdx.x;

    const float* src; int p;
    if (l == 0) { src = cls; p = 0; }
    else { src = iv + ((long)b * SEQ + (l - 1)) * HID; p = ip[b * SEQ + (l - 1)] + 1; }
    if (t == 0) posf[row] = p;

    float x[3]; float ls = 0.f;
    #pragma unroll
    for (int i = 0; i < 3; i++) { x[i] = src[t + i * 256]; ls += x[i]; }
    float mean = block_sum(ls, red) * (1.0f / HID);
    float lv = 0.f;
    #pragma unroll
    for (int i = 0; i < 3; i++) { float d = x[i] - mean; lv += d * d; }
    float inv = rsqrtf(block_sum(lv, red) * (1.0f / HID) + EPS);
    const float* pe = pos_emb + (long)p * HID;
    #pragma unroll
    for (int i = 0; i < 3; i++) {
        int c = t + i * 256;
        float y = (x[i] - mean) * inv * s[c] + bb[c] + pe[c];
        seq[(long)row * HID + c] = y;
        seqh[(long)row * HID + c] = (half_t)y;
    }
}

// ---------------- generic row LN (768 cols), f32+f16 out ----------------
__global__ __launch_bounds__(256) void k_ln_rows(
    const float* __restrict__ in, const float* __restrict__ s, const float* __restrict__ bb,
    float* __restrict__ out, half_t* __restrict__ outh)
{
    __shared__ float red[256];
    const long row = blockIdx.x; const int t = threadIdx.x;
    float x[3]; float ls = 0.f;
    #pragma unroll
    for (int i = 0; i < 3; i++) { x[i] = in[row * HID + t + i * 256]; ls += x[i]; }
    float mean = block_sum(ls, red) * (1.0f / HID);
    float lv = 0.f;
    #pragma unroll
    for (int i = 0; i < 3; i++) { float d = x[i] - mean; lv += d * d; }
    float inv = rsqrtf(block_sum(lv, red) * (1.0f / HID) + EPS);
    #pragma unroll
    for (int i = 0; i < 3; i++) {
        int c = t + i * 256;
        float y = (x[i] - mean) * inv * s[c] + bb[c];
        out[row * HID + c] = y;
        outh[row * HID + c] = (half_t)y;
    }
}

// ---------------- residual add + LN, f32+f16 out ----------------
__global__ __launch_bounds__(256) void k_add_ln(
    const float* __restrict__ x, const float* __restrict__ res,
    const float* __restrict__ s, const float* __restrict__ bb,
    float* __restrict__ out, half_t* __restrict__ outh)
{
    __shared__ float red[256];
    const long row = blockIdx.x; const int t = threadIdx.x;
    float v[3]; float ls = 0.f;
    #pragma unroll
    for (int i = 0; i < 3; i++) {
        long idx = row * HID + t + i * 256;
        v[i] = x[idx] + res[idx]; ls += v[i];
    }
    float mean = block_sum(ls, red) * (1.0f / HID);
    float lv = 0.f;
    #pragma unroll
    for (int i = 0; i < 3; i++) { float d = v[i] - mean; lv += d * d; }
    float inv = rsqrtf(block_sum(lv, red) * (1.0f / HID) + EPS);
    #pragma unroll
    for (int i = 0; i < 3; i++) {
        int c = t + i * 256;
        float y = (v[i] - mean) * inv * s[c] + bb[c];
        out[row * HID + c] = y;
        outh[row * HID + c] = (half_t)y;
    }
}

// ---------------- weight transpose + f32->f16: Wt[n*K+k] = W[k*N+n] ----------------
__global__ __launch_bounds__(256) void k_transpose_h(
    const float* __restrict__ W, half_t* __restrict__ Wt, int K, int N)
{
    __shared__ float tile[32][33];
    const int tx = threadIdx.x & 31, ty = threadIdx.x >> 5;   // 32x8
    const int n0 = blockIdx.x * 32, k0 = blockIdx.y * 32;
    #pragma unroll
    for (int i = 0; i < 4; i++)
        tile[ty + i * 8][tx] = W[(long)(k0 + ty + i * 8) * N + n0 + tx];
    __syncthreads();
    #pragma unroll
    for (int i = 0; i < 4; i++)
        Wt[(long)(n0 + ty + i * 8) * K + k0 + tx] = (half_t)tile[tx][ty + i * 8];
}

// ---------------- v transpose: vt[b][h][d][l] = v[b][l][h*64+d], f16 ----------------
__global__ __launch_bounds__(256) void k_vt(
    const float* __restrict__ v, half_t* __restrict__ vt, int total)
{
    int o = blockIdx.x * 256 + threadIdx.x;
    if (o >= total) return;
    int l = o & 511;
    int r = o >> 9;          // (b*12+h)*64 + d
    int d = r & 63;
    int hb = r >> 6;
    int h = hb % HEADS, b = hb / HEADS;
    vt[o] = (half_t)v[((long)b * LFULL + l) * HID + h * DH + d];
}

// ---------------- combine qk + gathered c2p/p2c, mask, softmax -> f16 probs ----------------
// buffers per batch b: [HEADS][512][512]; same gather index for c2p and p2c.
__global__ __launch_bounds__(256) void k_attn_combine(
    const float* __restrict__ qk, const float* __restrict__ c2p,
    const float* __restrict__ p2c, const int* __restrict__ posf,
    const int* __restrict__ am, half_t* __restrict__ probs, int b)
{
    __shared__ float red[256];
    const int h = blockIdx.x >> 9, l = blockIdx.x & 511;
    const int t = threadIdx.x;
    const int* pos = posf + b * LFULL;
    const int pl = pos[l];
    const int ml = (l == 0) ? 1 : am[b * SEQ + l - 1];

    float val[2]; float lmax = NEG_INF;
    #pragma unroll
    for (int i = 0; i < 2; i++) {
        int m = t + i * 256;
        int pm = pos[m];
        int mm = (m == 0) ? 1 : am[b * SEQ + m - 1];
        int idx = pl - pm + 256; idx = idx < 0 ? 0 : (idx > 511 ? 511 : idx);
        float v = (qk [((long)h * LFULL + l) * LFULL + m]
                 + c2p[((long)h * LFULL + l) * LFULL + idx]
                 + p2c[((long)h * LFULL + m) * LFULL + idx]) * INV_SCALE;
        bool keep = (ml > 0) && (mm > 0);
        val[i] = keep ? v : NEG_INF;
        lmax = fmaxf(lmax, val[i]);
    }
    float rmax = block_max(lmax, red);
    float lsum = 0.f;
    #pragma unroll
    for (int i = 0; i < 2; i++) { val[i] = __expf(val[i] - rmax); lsum += val[i]; }
    float rinv = 1.0f / block_sum(lsum, red);
    #pragma unroll
    for (int i = 0; i < 2; i++) {
        int m = t + i * 256;
        int mm = (m == 0) ? 1 : am[b * SEQ + m - 1];
        bool keep = (ml > 0) && (mm > 0);
        probs[((long)h * LFULL + l) * LFULL + m] = (half_t)(keep ? val[i] * rinv : 0.0f);
    }
}

// ---------------- host-side helpers ----------------
static inline void gemm(hipStream_t st, const half_t* A, const half_t* Bt, const float* bias,
                        float* C, half_t* C16, int M, int N, int K,
                        int lda, int ldb, int ldc, long sA, long sB, long sC,
                        int batches, int gelu, int bn)
{
    if (bn == 128) {
        dim3 g(N / 128, M / BM, batches);
        k_gemm<128><<<g, 256, 0, st>>>(A, Bt, bias, C, C16, M, N, K, lda, ldb, ldc, sA, sB, sC, gelu);
    } else {
        dim3 g(N / 64, M / BM, batches);
        k_gemm<64><<<g, 256, 0, st>>>(A, Bt, bias, C, C16, M, N, K, lda, ldb, ldc, sA, sB, sC, gelu);
    }
}

extern "C" void kernel_launch(void* const* d_in, const int* in_sizes, int n_in,
                              void* d_out, int out_size, void* d_ws, size_t ws_size,
                              hipStream_t stream)
{
    (void)in_sizes; (void)n_in; (void)out_size; (void)ws_size;

    const float* iv      = (const float*)d_in[0];
    const int*   ip      = (const int*)  d_in[1];
    const int*   am      = (const int*)  d_in[2];
    const float* cls     = (const float*)d_in[3];
    const float* pos_emb = (const float*)d_in[4];
    const float* emb_s   = (const float*)d_in[5];
    const float* emb_b   = (const float*)d_in[6];
    const float* rel_emb = (const float*)d_in[7];
    const float* rel_s   = (const float*)d_in[8];
    const float* rel_b   = (const float*)d_in[9];
    #define LP(layer, j) ((const float*)d_in[10 + (layer) * 20 + (j)])

    // ---- workspace bump allocator ----
    char* base = (char*)d_ws; size_t off = 0;
    auto alloc = [&](size_t bytes) -> void* {
        off = (off + 255) & ~(size_t)255;
        void* p = base + off; off += bytes; return p;
    };

    float*  seq    = (float*) alloc((size_t)ROWS * HID * 4);
    half_t* seqh   = (half_t*)alloc((size_t)ROWS * HID * 2);
    int*    posf   = (int*)   alloc((size_t)ROWS * 4);
    float*  relf   = (float*) alloc((size_t)LFULL * HID * 4);
    half_t* relh   = (half_t*)alloc((size_t)LFULL * HID * 2);
    half_t* wtq    = (half_t*)alloc((size_t)HID * HID * 2);
    half_t* wtk    = (half_t*)alloc((size_t)HID * HID * 2);
    half_t* wtv    = (half_t*)alloc((size_t)HID * HID * 2);
    half_t* wtpk   = (half_t*)alloc((size_t)HID * HID * 2);
    half_t* wtpq   = (half_t*)alloc((size_t)HID * HID * 2);
    half_t* wto    = (half_t*)alloc((size_t)HID * HID * 2);
    half_t* wti    = (half_t*)alloc((size_t)FFN * HID * 2);
    half_t* wto2   = (half_t*)alloc((size_t)HID * FFN * 2);
    float*  vf     = (float*) alloc((size_t)ROWS * HID * 4);
    half_t* qh     = (half_t*)alloc((size_t)ROWS * HID * 2);
    half_t* kh     = (half_t*)alloc((size_t)ROWS * HID * 2);
    half_t* vth    = (half_t*)alloc((size_t)BATCH * HEADS * DH * LFULL * 2);
    half_t* pkh    = (half_t*)alloc((size_t)LFULL * HID * 2);
    half_t* pqh    = (half_t*)alloc((size_t)LFULL * HID * 2);
    float*  qkb    = (float*) alloc((size_t)HEADS * LFULL * LFULL * 4);
    float*  c2pb   = (float*) alloc((size_t)HEADS * LFULL * LFULL * 4);
    float*  p2cb   = (float*) alloc((size_t)HEADS * LFULL * LFULL * 4);
    half_t* probsh = (half_t*)alloc((size_t)HEADS * LFULL * LFULL * 2);
    half_t* ctxh   = (half_t*)alloc((size_t)ROWS * HID * 2);
    float*  tmp1   = (float*) alloc((size_t)ROWS * HID * 4);
    half_t* interh = (half_t*)alloc((size_t)ROWS * FFN * 2);

    const long RH = (long)LFULL * HID;          // per-batch q/k row block
    const long HL2 = (long)LFULL * LFULL;       // per-head score block

    // ---- embeddings ----
    k_embed<<<ROWS, 256, 0, stream>>>(iv, ip, cls, pos_emb, emb_s, emb_b, seq, seqh, posf);
    k_ln_rows<<<LFULL, 256, 0, stream>>>(rel_emb, rel_s, rel_b, relf, relh);

    for (int layer = 0; layer < NL; layer++) {
        const float *wq = LP(layer,0),  *bq = LP(layer,1),  *wk = LP(layer,2),  *bk = LP(layer,3);
        const float *wv = LP(layer,4),  *bv = LP(layer,5),  *wpk= LP(layer,6),  *bpk= LP(layer,7);
        const float *wpq= LP(layer,8),  *bpq= LP(layer,9),  *wo = LP(layer,10), *bo = LP(layer,11);
        const float *l1s= LP(layer,12), *l1b= LP(layer,13), *wi = LP(layer,14), *bi = LP(layer,15);
        const float *wo2= LP(layer,16), *bo2= LP(layer,17), *l2s= LP(layer,18), *l2b= LP(layer,19);

        // weight transposes to f16 [N][K]
        dim3 t88(HID/32, HID/32);
        k_transpose_h<<<t88, 256, 0, stream>>>(wq,  wtq,  HID, HID);
        k_transpose_h<<<t88, 256, 0, stream>>>(wk,  wtk,  HID, HID);
        k_transpose_h<<<t88, 256, 0, stream>>>(wv,  wtv,  HID, HID);
        k_transpose_h<<<t88, 256, 0, stream>>>(wpk, wtpk, HID, HID);
        k_transpose_h<<<t88, 256, 0, stream>>>(wpq, wtpq, HID, HID);
        k_transpose_h<<<t88, 256, 0, stream>>>(wo,  wto,  HID, HID);
        k_transpose_h<<<dim3(FFN/32, HID/32), 256, 0, stream>>>(wi,  wti,  HID, FFN);
        k_transpose_h<<<dim3(HID/32, FFN/32), 256, 0, stream>>>(wo2, wto2, FFN, HID);

        // QKV projections
        gemm(stream, seqh, wtq, bq, nullptr, qh, ROWS, HID, HID, HID, HID, HID, 0,0,0, 1, 0, 128);
        gemm(stream, seqh, wtk, bk, nullptr, kh, ROWS, HID, HID, HID, HID, HID, 0,0,0, 1, 0, 128);
        gemm(stream, seqh, wtv, bv, vf, nullptr, ROWS, HID, HID, HID, HID, HID, 0,0,0, 1, 0, 128);
        k_vt<<<(BATCH*HEADS*DH*LFULL + 255)/256, 256, 0, stream>>>(vf, vth, BATCH*HEADS*DH*LFULL);

        // relative-position projections
        gemm(stream, relh, wtpk, bpk, nullptr, pkh, LFULL, HID, HID, HID, HID, HID, 0,0,0, 1, 0, 128);
        gemm(stream, relh, wtpq, bpq, nullptr, pqh, LFULL, HID, HID, HID, HID, HID, 0,0,0, 1, 0, 128);

        // attention per batch, heads batched in z
        for (int b = 0; b < BATCH; b++) {
            const half_t* qb = qh + (long)b * RH;
            const half_t* kb = kh + (long)b * RH;
            // scores = q @ k^T
            gemm(stream, qb, kb, nullptr, qkb, nullptr, LFULL, LFULL, DH,
                 HID, HID, LFULL, DH, DH, HL2, HEADS, 0, 128);
            // c2p = q @ pk^T
            gemm(stream, qb, pkh, nullptr, c2pb, nullptr, LFULL, LFULL, DH,
                 HID, HID, LFULL, DH, DH, HL2, HEADS, 0, 128);
            // p2c = k @ pq^T
            gemm(stream, kb, pqh, nullptr, p2cb, nullptr, LFULL, LFULL, DH,
                 HID, HID, LFULL, DH, DH, HL2, HEADS, 0, 128);
            // gather + mask + softmax
            k_attn_combine<<<HEADS * LFULL, 256, 0, stream>>>(qkb, c2pb, p2cb, posf, am, probsh, b);
            // ctx = probs @ v  (Bt = v^T per head), N=64 -> BN=64 variant
            gemm(stream, probsh, vth + (long)b * HEADS * DH * LFULL, nullptr,
                 nullptr, ctxh + (long)b * RH, LFULL, DH, LFULL,
                 LFULL, LFULL, HID, HL2, (long)DH * LFULL, DH, HEADS, 0, 64);
        }

        // output projection + residual LN
        gemm(stream, ctxh, wto, bo, tmp1, nullptr, ROWS, HID, HID, HID, HID, HID, 0,0,0, 1, 0, 128);
        k_add_ln<<<ROWS, 256, 0, stream>>>(tmp1, seq, l1s, l1b, seq, seqh);

        // FFN
        gemm(stream, seqh, wti, bi, nullptr, interh, ROWS, FFN, HID, HID, HID, FFN, 0,0,0, 1, 1, 128);
        gemm(stream, interh, wto2, bo2, tmp1, nullptr, ROWS, HID, FFN, FFN, FFN, HID, 0,0,0, 1, 0, 128);
        k_add_ln<<<ROWS, 256, 0, stream>>>(tmp1, seq, l2s, l2b, seq, seqh);
    }

    (void)hipMemcpyAsync(d_out, seq, (size_t)ROWS * HID * 4, hipMemcpyDeviceToDevice, stream);
}
